// Attention_90005334655189
// MI455X (gfx1250) — matrix operand order repaired and checked by reading, not compiled, not run on hardware
//
#include <hip/hip_runtime.h>

// ---------------------------------------------------------------------------
// Adaptive attention (sentinel gate) for MI455X / gfx1250, wave32 + WMMA bf16.
//
// B=64, T=512, K=49 (padded 64), H=2048.  fp32 in/out, bf16 WMMA internally.
// HBM floor ~47us (1.1GB @ 23.3TB/s); fp32 VALU would be >250us, so every
// contraction is routed through v_wmma_f32_16x16x32_bf16.
// Round 3: light LDS-only barrier (s_wait_dscnt + s_barrier_signal/wait -1)
// instead of __syncthreads so the global prefetch stays in flight across the
// barrier and overlaps the WMMA stream (avoids the s_wait_loadcnt 0 that the
// full-fence __syncthreads inserts).
// ---------------------------------------------------------------------------

#define BATCH 64
#define TLEN  512
#define KDIM  49
#define KPAD  64
#define HID   2048
#define ROWS  (BATCH * TLEN)   // 32768

typedef __attribute__((ext_vector_type(8)))  unsigned short v8us;
typedef __attribute__((ext_vector_type(16))) unsigned short v16us;
typedef __attribute__((ext_vector_type(16))) __bf16         v16bf;
typedef __attribute__((ext_vector_type(8)))  float          v8f;
typedef __attribute__((ext_vector_type(4)))  unsigned int   v4ui;

// Workgroup barrier with LDS-only visibility: waits DScnt (LDS stores visible,
// this wave's LDS fragment reads drained) but does NOT wait LOADcnt, so
// register-destined global prefetches remain outstanding across the barrier.
__device__ __forceinline__ void wg_barrier_lds() {
    asm volatile("s_wait_dscnt 0x0\n\t"
                 "s_barrier_signal -1\n\t"
                 "s_barrier_wait -1" ::: "memory");
}

__device__ __forceinline__ unsigned short f2bf(float f) {
    unsigned int u = __builtin_bit_cast(unsigned int, f);
    unsigned int r = u + 0x7fffu + ((u >> 16) & 1u);   // round-to-nearest-even
    return (unsigned short)(r >> 16);
}

#if __has_builtin(__builtin_amdgcn_cvt_pk_bf16_f32)
// hardware packed convert: v_cvt_pk_bf16_f32 (src0 -> low, src1 -> high)
__device__ __forceinline__ v8us cvt8(float4 a, float4 b) {
    v4ui t;
    t.x = __builtin_bit_cast(unsigned int, __builtin_amdgcn_cvt_pk_bf16_f32(a.x, a.y));
    t.y = __builtin_bit_cast(unsigned int, __builtin_amdgcn_cvt_pk_bf16_f32(a.z, a.w));
    t.z = __builtin_bit_cast(unsigned int, __builtin_amdgcn_cvt_pk_bf16_f32(b.x, b.y));
    t.w = __builtin_bit_cast(unsigned int, __builtin_amdgcn_cvt_pk_bf16_f32(b.z, b.w));
    return __builtin_bit_cast(v8us, t);
}
#else
__device__ __forceinline__ v8us cvt8(float4 a, float4 b) {
    v8us r;
    r[0] = f2bf(a.x); r[1] = f2bf(a.y); r[2] = f2bf(a.z); r[3] = f2bf(a.w);
    r[4] = f2bf(b.x); r[5] = f2bf(b.y); r[6] = f2bf(b.z); r[7] = f2bf(b.w);
    return r;
}
#endif

// Load a 16-element bf16 fragment as two 16B chunks at p and p+secondOff.
__device__ __forceinline__ v16bf ld_frag(const unsigned short* p, int secondOff) {
    v8us a = *reinterpret_cast<const v8us*>(p);
    v8us b = *reinterpret_cast<const v8us*>(p + secondOff);
    v16us c = __builtin_shufflevector(a, b, 0,1,2,3,4,5,6,7,8,9,10,11,12,13,14,15);
    return __builtin_bit_cast(v16bf, c);
}

__device__ __forceinline__ v16bf cat_frag(v8us a, v8us b) {
    v16us c = __builtin_shufflevector(a, b, 0,1,2,3,4,5,6,7,8,9,10,11,12,13,14,15);
    return __builtin_bit_cast(v16bf, c);
}

__device__ __forceinline__ float tanh_fast(float x) {
    x = fminf(fmaxf(x, -15.f), 15.f);
    float e = __expf(2.f * x);           // v_exp_f32 path
    return (e - 1.f) / (e + 1.f);
}

// ---------------------------------------------------------------------------
// Kernel 0: convert the three (49,2048) fp32 weights to bf16, zero-padded to
// 64 rows (so padded GEMM columns are exactly zero).
// ---------------------------------------------------------------------------
__global__ __launch_bounds__(256)
void cvt_weights(const float* __restrict__ Wv, const float* __restrict__ Wg,
                 const float* __restrict__ Ws,
                 unsigned short* __restrict__ WvB, unsigned short* __restrict__ WgB,
                 unsigned short* __restrict__ WsB) {
    int gid = blockIdx.x * 256 + threadIdx.x;     // 3 * 64 * 2048 total
    int w   = gid >> 17;                          // 131072 per weight
    int idx = gid & 131071;
    int r = idx >> 11, d = idx & 2047;
    const float* src      = (w == 0) ? Wg : (w == 1) ? Ws : Wv;
    unsigned short* dst   = (w == 0) ? WgB : (w == 1) ? WsB : WvB;
    float v = (r < KDIM) ? src[r * HID + d] : 0.f;
    dst[idx] = f2bf(v);
}

// ---------------------------------------------------------------------------
// Kernel 1: generic OUT[m][n] = sum_d A[m][d] * W[n][d], N = 64, per-block
// M-tile = 64, K-step 32.  4 waves, each owns a 16-col slice and 4 row tiles.
// Double-buffered LDS A tile (pitch 40 ushorts = 80B: 16B aligned + bank-
// conflict-free fragment loads); next iteration's global loads issued before
// the light barrier so HBM latency overlaps barrier + WMMA.  One barrier/iter
// is sufficient: it waits DScnt, so readers of buffer X at iter i are fenced
// by the barrier of iter i+1 before X is overwritten at iter i+2.
// Used for: pg = h_t*Wg^T, ps = sent_t*Ws^T, pv = V_b*Wv^T (rows guarded).
// ---------------------------------------------------------------------------
__global__ __launch_bounds__(128)
void gemm_nt_bf16(const float* __restrict__ A, const unsigned short* __restrict__ Wb,
                  float* __restrict__ Out, int blockRowAdvance, int validRows) {
    __shared__ unsigned short aLds[2][64 * 40];

    const int tid  = threadIdx.x;
    const int lane = tid & 31;
    const int wave = tid >> 5;
    const long rowBase = (long)blockIdx.x * blockRowAdvance;

    v8f acc[4] = {};

    const int  r   = tid >> 1;            // staged row 0..63
    const int  cs  = (tid & 1) * 16;      // staged col segment
    const bool rOk = (r < validRows);
    const float* aRow = A + (rowBase + r) * (long)HID;

    const int nCol = wave * 16 + (lane & 15);            // B column (= W row)
    const int kb8  = (lane >> 4) * 8;                    // A-fragment K base
    const unsigned short* wRow = Wb + (long)nCol * HID + (lane >> 4) * 8;

    // prologue loads (kt = 0)
    float4 a0{}, a1{}, a2{}, a3{};
    if (rOk) {
        const float4* p = reinterpret_cast<const float4*>(aRow + cs);
        a0 = p[0]; a1 = p[1]; a2 = p[2]; a3 = p[3];
    }
    v8us b0 = *reinterpret_cast<const v8us*>(wRow);
    v8us b1 = *reinterpret_cast<const v8us*>(wRow + 16);

    for (int kt = 0; kt < HID / 32; ++kt) {
        const int buf = kt & 1;
        // stage current A tile (fp32 regs -> bf16 LDS)
        *reinterpret_cast<v8us*>(&aLds[buf][r * 40 + cs])     = cvt8(a0, a1);
        *reinterpret_cast<v8us*>(&aLds[buf][r * 40 + cs + 8]) = cvt8(a2, a3);

        // prefetch next tile (A rows + B fragment) before the barrier;
        // these loads stay in flight across the light barrier.
        float4 n0 = a0, n1 = a1, n2 = a2, n3 = a3;
        v8us  nb0 = b0, nb1 = b1;
        if (kt + 1 < HID / 32) {
            const int k1 = (kt + 1) * 32;
            if (rOk) {
                const float4* p = reinterpret_cast<const float4*>(aRow + k1 + cs);
                n0 = p[0]; n1 = p[1]; n2 = p[2]; n3 = p[3];
            }
            nb0 = *reinterpret_cast<const v8us*>(wRow + k1);
            nb1 = *reinterpret_cast<const v8us*>(wRow + k1 + 16);
        }
        wg_barrier_lds();

        v16bf bF = cat_frag(b0, b1);
#pragma unroll
        for (int mt = 0; mt < 4; ++mt) {
            const unsigned short* ap = &aLds[buf][(mt * 16 + (lane & 15)) * 40 + kb8];
            v16bf aF = ld_frag(ap, 16);
            acc[mt] = __builtin_amdgcn_wmma_f32_16x16x32_bf16(
                false, aF, false, bF, (short)0, acc[mt], false, false);
        }
        a0 = n0; a1 = n1; a2 = n2; a3 = n3;
        b0 = nb0; b1 = nb1;
    }

    float* outB = Out + (long)blockIdx.x * 64 * 64;
    const int rHalf = (lane >> 4) * 8;
#pragma unroll
    for (int mt = 0; mt < 4; ++mt)
#pragma unroll
        for (int j = 0; j < 8; ++j)
            outB[(mt * 16 + rHalf + j) * 64 + nCol] = acc[mt][j];
}

// ---------------------------------------------------------------------------
// Kernel 2: per (b,t):  z_t[k] = sum_j tanh(pv[b,k,j]+pg[b,t,j])*wh[j]
//                       z_ext  = sum_j tanh(ps[b,t,j]+pg[b,t,j])*wh[j]
// alpha = softmax_49(z_t); beta = last slot of softmax_50([z_t, z_ext]).
// Block = 256 threads = 4 t-groups of 64 (k = lane-in-group; k==49 -> z_ext).
// pv tile lives in LDS with pitch 53 (odd -> conflict-free column walks).
// ---------------------------------------------------------------------------
__global__ __launch_bounds__(256)
void attn_softmax(const float* __restrict__ pv, const float* __restrict__ pg,
                  const float* __restrict__ ps, const float* __restrict__ wh,
                  float* __restrict__ alphaOut, float* __restrict__ betaOut,
                  float* __restrict__ alphaPad) {
    __shared__ float pvL[KDIM * 53];
    __shared__ float whL[64];
    __shared__ float pgL[4][52];
    __shared__ float psL[4][52];
    __shared__ float zL[4][64];

    const int tid    = threadIdx.x;
    const int btBase = blockIdx.x * 4;
    const int b      = btBase >> 9;          // T = 512
    const int tg     = tid >> 6;
    const int kk     = tid & 63;

    const float* pvB = pv + (long)b * (KPAD * KPAD);
    for (int i = tid; i < KDIM * KDIM; i += 256) {
        int k = i / KDIM, j = i - k * KDIM;
        pvL[k * 53 + j] = pvB[k * KPAD + j];
    }
    if (tid < 64) whL[tid] = (tid < KDIM) ? wh[tid] : 0.f;
    if (kk < KDIM) {
        pgL[tg][kk] = pg[(long)(btBase + tg) * KPAD + kk];
        psL[tg][kk] = ps[(long)(btBase + tg) * KPAD + kk];
    }
    wg_barrier_lds();

    if (kk < KDIM) {
        float acc = 0.f;
#pragma unroll 7
        for (int j = 0; j < KDIM; ++j)
            acc += tanh_fast(pvL[kk * 53 + j] + pgL[tg][j]) * whL[j];
        zL[tg][kk] = acc;
    } else if (kk == KDIM) {
        float acc = 0.f;
#pragma unroll 7
        for (int j = 0; j < KDIM; ++j)
            acc += tanh_fast(psL[tg][j] + pgL[tg][j]) * whL[j];
        zL[tg][KDIM] = acc;
    }
    wg_barrier_lds();

    // redundant-but-uniform reductions (LDS broadcast reads)
    float m49 = -1e30f;
    for (int j = 0; j < KDIM; ++j) m49 = fmaxf(m49, zL[tg][j]);
    float s49 = 0.f;
    for (int j = 0; j < KDIM; ++j) s49 += __expf(zL[tg][j] - m49);
    const float ze = zL[tg][KDIM];
    float m50, s50;
    if (ze <= m49) { m50 = m49; s50 = s49 + __expf(ze - m49); }
    else           { m50 = ze;  s50 = s49 * __expf(m49 - ze) + 1.f; }

    const long bt = btBase + tg;
    if (kk < KDIM) {
        float a = __expf(zL[tg][kk] - m49) / s49;
        alphaOut[bt * KDIM + kk] = a;
        alphaPad[bt * KPAD + kk] = a;
    } else {
        alphaPad[bt * KPAD + kk] = 0.f;       // zero pad -> no K guard in ct GEMM
        if (kk == KDIM) betaOut[bt] = __expf(ze - m50) / s50;
    }
}

// ---------------------------------------------------------------------------
// Kernel 3: c_t = alpha * V (per b: (512x64pad)@(64pad x 2048)) via WMMA,
// fused epilogue c_hat = beta*sent + (1-beta)*c_t.
// Block = 256 thr (8 waves): M-tile 64 t, N-tile 128 d, K = 64 (2 phases).
// All global loads issued up front; two LDS buffers -> one light barrier per
// phase.  V is staged fp32->bf16 TRANSPOSED in LDS ([d][k], pitch 40) so
// B-fragment columns are two contiguous 16B LDS reads.
// ---------------------------------------------------------------------------
__global__ __launch_bounds__(256)
void wmma_ct(const float* __restrict__ V, const float* __restrict__ sent,
             const float* __restrict__ alphaPad, const float* __restrict__ beta,
             float* __restrict__ out) {
    __shared__ unsigned short aLds[2][64 * 40];    // alpha: [t][k]
    __shared__ unsigned short bLds[2][128 * 40];   // V^T:   [d][k]
    __shared__ float betaL[64];

    const int tid  = threadIdx.x;
    const int lane = tid & 31;
    const int wave = tid >> 5;
    const int dBase = blockIdx.x * 128;
    const int tBase = blockIdx.y * 64;
    const int b     = blockIdx.z;
    const long btBase = (long)b * TLEN + tBase;

    if (tid < 64) betaL[tid] = beta[btBase + tid];

    v8f acc[4] = {};

    const int ar = tid >> 2, acs = (tid & 3) * 8;      // A staging: row, 8-col seg
    const int bk = tid >> 3, bds = (tid & 7) * 16;     // B staging: k row, 16-d seg
    const int kb8 = (lane >> 4) * 8;

    // front-load both K phases (k0 = 0, 32) into registers
    float4 aA[2][2];
    float4 bB[2][4];
#pragma unroll
    for (int kt = 0; kt < 2; ++kt) {
        const int k0 = kt * 32;
        const float4* pa = reinterpret_cast<const float4*>(
            alphaPad + (btBase + ar) * KPAD + k0 + acs);
        aA[kt][0] = pa[0]; aA[kt][1] = pa[1];
        const int kg = k0 + bk;
        float4 f0{}, f1{}, f2{}, f3{};
        if (kg < KDIM) {
            const float4* pb = reinterpret_cast<const float4*>(
                V + ((long)b * KDIM + kg) * HID + dBase + bds);
            f0 = pb[0]; f1 = pb[1]; f2 = pb[2]; f3 = pb[3];
        }
        bB[kt][0] = f0; bB[kt][1] = f1; bB[kt][2] = f2; bB[kt][3] = f3;
    }

#pragma unroll
    for (int kt = 0; kt < 2; ++kt) {
        // stage alpha tile 64x32 (already zero-padded past k=48)
        *reinterpret_cast<v8us*>(&aLds[kt][ar * 40 + acs]) = cvt8(aA[kt][0], aA[kt][1]);
        // stage V chunk 32k x 128d, transposed into LDS
        {
            float4 f0 = bB[kt][0], f1 = bB[kt][1], f2 = bB[kt][2], f3 = bB[kt][3];
            float vals[16] = { f0.x,f0.y,f0.z,f0.w, f1.x,f1.y,f1.z,f1.w,
                               f2.x,f2.y,f2.z,f2.w, f3.x,f3.y,f3.z,f3.w };
#pragma unroll
            for (int i = 0; i < 16; ++i)
                bLds[kt][(bds + i) * 40 + bk] = f2bf(vals[i]);
        }
        wg_barrier_lds();

        const unsigned short* bp =
            &bLds[kt][(wave * 16 + (lane & 15)) * 40 + (lane >> 4) * 8];
        v16bf bF = ld_frag(bp, 16);
#pragma unroll
        for (int mt = 0; mt < 4; ++mt) {
            const unsigned short* ap = &aLds[kt][(mt * 16 + (lane & 15)) * 40 + kb8];
            v16bf aF = ld_frag(ap, 16);
            acc[mt] = __builtin_amdgcn_wmma_f32_16x16x32_bf16(
                false, aF, false, bF, (short)0, acc[mt], false, false);
        }
    }

    // epilogue: c_hat = beta*sent + (1-beta)*c_t
    const int colD  = dBase + wave * 16 + (lane & 15);
    const int rHalf = (lane >> 4) * 8;
#pragma unroll
    for (int mt = 0; mt < 4; ++mt)
#pragma unroll
        for (int j = 0; j < 8; ++j) {
            const int  rLoc = mt * 16 + rHalf + j;
            const long bt   = btBase + rLoc;
            const float bv  = betaL[rLoc];
            const float s   = sent[bt * HID + colD];
            out[bt * HID + colD] = bv * s + (1.f - bv) * acc[mt][j];
        }
}

// ---------------------------------------------------------------------------
extern "C" void kernel_launch(void* const* d_in, const int* in_sizes, int n_in,
                              void* d_out, int out_size, void* d_ws, size_t ws_size,
                              hipStream_t stream) {
    (void)in_sizes; (void)n_in; (void)out_size; (void)ws_size;
    const float* V    = (const float*)d_in[0];
    const float* h_t  = (const float*)d_in[1];
    const float* sent = (const float*)d_in[2];
    const float* Wv   = (const float*)d_in[3];
    const float* Wg   = (const float*)d_in[4];
    const float* Ws   = (const float*)d_in[5];
    const float* Wh   = (const float*)d_in[6];

    // workspace layout
    const size_t WELT = (size_t)KPAD * HID;                 // ushorts per weight
    unsigned short* WgB = (unsigned short*)d_ws;
    unsigned short* WsB = WgB + WELT;
    unsigned short* WvB = WsB + WELT;
    float* pgW      = (float*)(WvB + WELT);                 // ROWS x 64
    float* psW      = pgW + (size_t)ROWS * KPAD;            // ROWS x 64
    float* pvW      = psW + (size_t)ROWS * KPAD;            // B x 64 x 64
    float* alphaPad = pvW + (size_t)BATCH * KPAD * KPAD;    // ROWS x 64

    // output layout: c_hat (B*T*H) | alpha (B*T*49) | beta (B*T)
    float* cOut = (float*)d_out;
    float* aOut = cOut + (size_t)ROWS * HID;
    float* bOut = aOut + (size_t)ROWS * KDIM;

    cvt_weights<<<(3 * KPAD * HID) / 256, 256, 0, stream>>>(Wv, Wg, Ws, WvB, WgB, WsB);

    gemm_nt_bf16<<<ROWS / 64, 128, 0, stream>>>(h_t,  WgB, pgW, 64, 64);
    gemm_nt_bf16<<<ROWS / 64, 128, 0, stream>>>(sent, WsB, psW, 64, 64);
    gemm_nt_bf16<<<BATCH,     128, 0, stream>>>(V,    WvB, pvW, KDIM, KDIM);

    attn_softmax<<<ROWS / 4, 256, 0, stream>>>(pvW, pgW, psW, Wh, aOut, bOut, alphaPad);

    wmma_ct<<<dim3(HID / 128, TLEN / 64, BATCH), 256, 0, stream>>>(
        V, sent, alphaPad, bOut, cOut);
}
